// SSM_35751307771912
// MI455X (gfx1250) — matrix operand order repaired and checked by reading, not compiled, not run on hardware
//
#include <hip/hip_runtime.h>

// S4D chunked-scan reformulation for gfx1250 (MI455X).
// Grid: 1024 blocks (one per hidden channel h). Block: 128 threads = 4 wave32.
// Per h: 3 GEMMs of 64x128x64 via v_wmma_f32_16x16x32_bf16 + a 32-step scan.

#define Hdim 1024
#define Ldim 2048
#define Bdim 4
#define N2m  32
#define NS   64     // real state dims (2*N2)
#define Tc   64     // chunk length
#define NCh  32     // number of chunks
#define NCOL 128    // Bdim * NCh columns

typedef __bf16 bf16;
typedef bf16  v16bf __attribute__((ext_vector_type(16)));
typedef float v8f   __attribute__((ext_vector_type(8)));

#define SA  66      // padded bf16 row stride (avoid LDS bank conflicts)
#define SBS 129     // padded f32 stride for Bsum

union FragU { v16bf v; unsigned u[8]; };

// A-operand fragment: 16x32 bf16 tile from row-major [row][K] LDS matrix.
// ISA 7.12.2: lane L holds row M=L%16; VGPR g: K = (g>=4?16:0) + (L>=16?8:0) + 2*(g&3) + {0,1}
__device__ __forceinline__ v16bf load_a(const bf16* base, int row0, int k0, int lane) {
  FragU f;
  const bf16* p = base + (row0 + (lane & 15)) * SA + k0 + ((lane >> 4) << 3);
  #pragma unroll
  for (int g = 0; g < 8; ++g) {
    const int kb = ((g >= 4) ? 16 : 0) + ((g & 3) << 1);
    f.u[g] = *(const unsigned*)(p + kb);   // dword = two consecutive K values
  }
  return f.v;
}

// B-operand fragment: 32x16 bf16 tile, from TRANSPOSED [col][K] LDS storage.
// ISA: lane L holds col N=L%16; VGPR v: K = (v>=4?16:0) + (L>=16?8:0) + 2*(v&3) + {0,1}
__device__ __forceinline__ v16bf load_bt(const bf16* baseT, int col0, int k0, int lane) {
  FragU f;
  const bf16* p = baseT + (col0 + (lane & 15)) * SA + k0 + ((lane >> 4) << 3);
  #pragma unroll
  for (int v = 0; v < 8; ++v) f.u[v] = *(const unsigned*)(p + ((v >= 4) ? 16 : 0) + ((v & 3) << 1));
  return f.v;
}

__global__ __launch_bounds__(128)
void s4d_chunked_wmma(const float* __restrict__ inp,    // [B,L,H]
                      const float* __restrict__ logdt,  // [H]
                      const float* __restrict__ Dp,     // [H]
                      const float* __restrict__ Alog,   // [H,N2]
                      const float* __restrict__ Aim,    // [H,N2]
                      const float* __restrict__ Bp,     // [H,N2,2]
                      const float* __restrict__ Cp,     // [H,N2,2]
                      float* __restrict__ outp)         // [B,L,H]
{
  __shared__ __align__(16) bf16  sUT  [NCOL * SA];  // U^T:  [col][s]
  __shared__ __align__(16) bf16  sX0T [NCOL * SA];  // X0^T: [col][j]
  __shared__ __align__(16) bf16  sKtri[Tc * SA];    // [t][s] causal Toeplitz
  __shared__ __align__(16) bf16  sWout[Tc * SA];    // [t][j]
  __shared__ __align__(16) bf16  sBmat[NS * SA];    // [j][s]
  __shared__ __align__(16) float sBsum[NS * SBS];   // [j][col] f32 chunk summaries
  __shared__ float kvec[Tc];
  __shared__ float sATr[N2m], sATi[N2m];            // A_disc^Tc

  const int tid  = threadIdx.x;
  const int h    = blockIdx.x;
  const int lane = tid & 31;
  const int wave = tid >> 5;

  if (tid < Tc) kvec[tid] = 0.f;

  // ---- Stage U into LDS as bf16, transposed: col = b*32 + c, inner = s
  {
    const int col = tid;
    const int b = col >> 5, c = col & 31;
    const float* up = inp + ((size_t)b * Ldim + (size_t)c * Tc) * Hdim + h;
    #pragma unroll 4
    for (int s = 0; s < Tc; ++s)
      sUT[col * SA + s] = (bf16)up[(size_t)s * Hdim];
  }
  __syncthreads();

  // ---- Discretize (bilinear) + generate powers of A_disc; build Wout/Bmat/k
  if (tid < N2m) {
    const int n = tid;
    const float dt  = __expf(logdt[h]);
    const float ar  = -__expf(Alog[h * N2m + n]);
    const float ai  = Aim[h * N2m + n];
    const float hdt = 0.5f * dt;
    const float denr = 1.f - hdt * ar, deni = -hdt * ai;
    const float numr = 1.f + hdt * ar, numi =  hdt * ai;
    const float inv  = 1.f / (denr * denr + deni * deni);
    const float adr = (numr * denr + numi * deni) * inv;      // A_disc
    const float adi = (numi * denr - numr * deni) * inv;
    const float b0r = Bp[(h * N2m + n) * 2 + 0], b0i = Bp[(h * N2m + n) * 2 + 1];
    const float bdr = dt * (b0r * denr + b0i * deni) * inv;   // B_disc
    const float bdi = dt * (b0i * denr - b0r * deni) * inv;
    const float cr  = 2.f * Cp[(h * N2m + n) * 2 + 0];        // 2*C
    const float ci  = 2.f * Cp[(h * N2m + n) * 2 + 1];
    const float cbr = cr * bdr - ci * bdi;                    // 2C * B_disc
    const float cbi = cr * bdi + ci * bdr;
    float pr = 1.f, pi = 0.f;                                 // p = A_disc^tau
    for (int tau = 0; tau < Tc; ++tau) {
      atomicAdd(&kvec[tau], cbr * pr - cbi * pi);             // k[tau] = Re(2C B A^tau)
      sBmat[(2 * n    ) * SA + (Tc - 1 - tau)] = (bf16)(pr * bdr - pi * bdi);
      sBmat[(2 * n + 1) * SA + (Tc - 1 - tau)] = (bf16)(pr * bdi + pi * bdr);
      const float npr = pr * adr - pi * adi;
      const float npi = pr * adi + pi * adr;
      pr = npr; pi = npi;                                     // now A^{tau+1}
      sWout[tau * SA + 2 * n    ] = (bf16)( cr * pr - ci * pi);
      sWout[tau * SA + 2 * n + 1] = (bf16)(-(cr * pi + ci * pr));
    }
    sATr[n] = pr; sATi[n] = pi;                               // A_disc^64
  }
  __syncthreads();

  // ---- Build causal Toeplitz Ktri[t][s] = k[t-s]
  for (int idx = tid; idx < Tc * Tc; idx += 128) {
    const int t = idx >> 6, s = idx & 63;
    sKtri[t * SA + s] = (s <= t) ? (bf16)kvec[t - s] : (bf16)0.f;
  }
  __syncthreads();

  // ---- GEMM1: Bsum(64x128) = Bmat(64x64) @ U(64x128), f32 accumulate
  #pragma unroll
  for (int cti = 0; cti < 2; ++cti) {
    const int ct = wave * 2 + cti;
    #pragma unroll
    for (int rt = 0; rt < 4; ++rt) {
      v8f acc = {0.f, 0.f, 0.f, 0.f, 0.f, 0.f, 0.f, 0.f};
      #pragma unroll
      for (int k0 = 0; k0 < NS; k0 += 32) {
        v16bf fa = load_a (sBmat, rt * 16, k0, lane);
        v16bf fb = load_bt(sUT,   ct * 16, k0, lane);
        acc = __builtin_amdgcn_wmma_f32_16x16x32_bf16(false, fa, false, fb,
                                                      (short)0, acc, false, false);
      }
      const int rowb = rt * 16 + ((lane >= 16) ? 8 : 0);
      const int colb = ct * 16 + (lane & 15);
      #pragma unroll
      for (int r = 0; r < 8; ++r)
        sBsum[(rowb + r) * SBS + colb] = acc[r];
    }
  }
  __syncthreads();

  // ---- Inter-chunk scan (only 32 serial steps): thread = (mode n, batch b)
  {
    const int n = tid & 31, b = tid >> 5;
    const float atr = sATr[n], ati = sATi[n];
    float xr = 0.f, xi = 0.f;
    for (int c = 0; c < NCh; ++c) {
      const int col = b * NCh + c;
      sX0T[col * SA + 2 * n    ] = (bf16)xr;   // state ENTERING chunk c
      sX0T[col * SA + 2 * n + 1] = (bf16)xi;
      const float br = sBsum[(2 * n    ) * SBS + col];
      const float bi = sBsum[(2 * n + 1) * SBS + col];
      const float nr = atr * xr - ati * xi + br;
      const float ni = atr * xi + ati * xr + bi;
      xr = nr; xi = ni;
    }
  }
  __syncthreads();

  // ---- GEMM2: Y = Ktri@U + Wout@X0, epilogue adds D*u, scatter to out
  const float Dh = Dp[h];
  #pragma unroll
  for (int cti = 0; cti < 2; ++cti) {
    const int ct = wave * 2 + cti;
    #pragma unroll
    for (int rt = 0; rt < 4; ++rt) {
      v8f acc = {0.f, 0.f, 0.f, 0.f, 0.f, 0.f, 0.f, 0.f};
      #pragma unroll
      for (int k0 = 0; k0 < Tc; k0 += 32) {
        v16bf fa = load_a (sKtri, rt * 16, k0, lane);
        v16bf fb = load_bt(sUT,   ct * 16, k0, lane);
        acc = __builtin_amdgcn_wmma_f32_16x16x32_bf16(false, fa, false, fb,
                                                      (short)0, acc, false, false);
      }
      #pragma unroll
      for (int k0 = 0; k0 < NS; k0 += 32) {
        v16bf fa = load_a (sWout, rt * 16, k0, lane);
        v16bf fb = load_bt(sX0T,  ct * 16, k0, lane);
        acc = __builtin_amdgcn_wmma_f32_16x16x32_bf16(false, fa, false, fb,
                                                      (short)0, acc, false, false);
      }
      const int coln = ct * 16 + (lane & 15);
      const int bb = coln >> 5, cc = coln & 31;
      const int rbase = rt * 16 + ((lane >= 16) ? 8 : 0);
      #pragma unroll
      for (int r = 0; r < 8; ++r) {
        const int s = rbase + r;
        const int l = cc * Tc + s;
        const size_t gi = ((size_t)bb * Ldim + l) * Hdim + h;
        outp[gi] = acc[r] + Dh * inp[gi];   // D*u term in full f32 (reload via L2)
      }
    }
  }
}

extern "C" void kernel_launch(void* const* d_in, const int* in_sizes, int n_in,
                              void* d_out, int out_size, void* d_ws, size_t ws_size,
                              hipStream_t stream) {
  (void)in_sizes; (void)n_in; (void)d_ws; (void)ws_size; (void)out_size;
  const float* inp   = (const float*)d_in[0];
  const float* logdt = (const float*)d_in[1];
  const float* Dp    = (const float*)d_in[2];
  const float* Alog  = (const float*)d_in[3];
  const float* Aim   = (const float*)d_in[4];
  const float* Bp    = (const float*)d_in[5];
  const float* Cp    = (const float*)d_in[6];
  float* outp = (float*)d_out;
  s4d_chunked_wmma<<<dim3(Hdim), dim3(128), 0, stream>>>(
      inp, logdt, Dp, Alog, Aim, Bp, Cp, outp);
}
